// Custom_22488448761921
// MI455X (gfx1250) — compile-verified, matrix operand order checked by reading, not run on hardware
//
#include <hip/hip_runtime.h>
#include <hip/hip_bf16.h>
#include <float.h>

typedef __attribute__((ext_vector_type(16))) _Float16 v16h;
typedef __attribute__((ext_vector_type(8)))  float    v8f;

#define L_SEQ   50000
#define E_EDGES 1600000
#define CIN     20
#define HID     64
#define C2      128
#define LP      (L_SEQ + 2)          // padded hidden length (1 zero each side)
#define LOUT    25000                // conv output length
#define TILES_A (L_SEQ / 16)         // 3125
#define TILES_C ((LOUT + 15) / 16)   // 1563

// ---- workspace layout (bytes) ----
#define WS_ACC   0u                  // accT: [L][20] f32  = 4,000,000 (16,000,000 reserved)
#define WS_CNT   16000000u           // cnt_src: [L] f32   = 200,000
#define WS_DEG   16200000u           // deg(dst): [L] f32  = 200,000
#define WS_H     16400000u           // h: [64][LP] f16    = 6,400,256 (pad to 6,400,512)
#define WS_XT    22800512u           // xT: [L][20] f32    = 4,000,000 (16,000,000 reserved)
#define WS_A1    38800512u           // A1: [64][64] f16   = 8,192
#define WS_A2    38808704u           // A2: [128][192] f16 = 49,152
#define WS_PART  38857856u           // partials: [TILES_C][128] f32 = 800,256
#define WS_ZERO_DWORDS (WS_XT / 4)   // zero accT+cnt+deg+h in one pass

// ---------------- zero scratch ----------------
__global__ void k_zero(unsigned* p, int n) {
    int i = blockIdx.x * blockDim.x + threadIdx.x;
    int stride = gridDim.x * blockDim.x;
    for (; i < n; i += stride) p[i] = 0u;
}

// ---------------- transpose x: (20,L) -> (L,20) ----------------
__global__ void k_transpose(const float* __restrict__ seq, float* __restrict__ xT) {
    int i = blockIdx.x * blockDim.x + threadIdx.x;
    if (i >= CIN * L_SEQ) return;
    int l = i / CIN, c = i - l * CIN;
    xT[i] = seq[c * L_SEQ + l];
}

// ---------------- pack f16 A-matrices ----------------
// A1 (64x64): rows 0..62 = [W0[o,0:20] | W1[o,0:20] | 0...], row 63 = 0
// A2 (128x192): k = r*64 + ic  ->  w_conv[oc][ic][r]
__global__ void k_prep(const float* __restrict__ w_pair, const float* __restrict__ w_conv,
                       _Float16* __restrict__ A1, _Float16* __restrict__ A2) {
    int i = blockIdx.x * blockDim.x + threadIdx.x;
    if (i < 64 * 64) {
        int o = i >> 6, k = i & 63;
        float v = 0.f;
        if (o < 63) {
            if (k < 20)      v = w_pair[(o * 20 + k) * 2 + 0];
            else if (k < 40) v = w_pair[(o * 20 + (k - 20)) * 2 + 1];
        }
        A1[i] = (_Float16)v;
    } else {
        int j = i - 64 * 64;
        if (j < 128 * 192) {
            int oc = j / 192, k = j - oc * 192;
            int r = k >> 6, ic = k & 63;
            A2[j] = (_Float16)w_conv[(oc * 64 + ic) * 3 + r];
        }
    }
}

// ---------------- edge scatter: acc[:,src] += x[:,dst]; cnt_src; deg(dst) ----------------
__global__ void k_edge(const int* __restrict__ src, const int* __restrict__ dst,
                       const float* __restrict__ xT, float* __restrict__ accT,
                       float* __restrict__ cnt, float* __restrict__ deg) {
    int idx = blockIdx.x * blockDim.x + threadIdx.x;
    if (idx >= E_EDGES * CIN) return;
    int e = idx / CIN;
    int c = idx - e * CIN;
    int s = src[e];
    int d = dst[e];
    atomicAdd(&accT[s * CIN + c], xT[d * CIN + c]);
    if (c == 0)      atomicAdd(&cnt[s], 1.0f);
    else if (c == 1) atomicAdd(&deg[d], 1.0f);
}

// ---------------- stage A: h = relu((A1 * [cnt*x ; acc]) * rdeg), ch63 = deg ----------------
__device__ __forceinline__ float bvalA(int k, const float* __restrict__ xT,
                                       const float* __restrict__ accT, int l, float cnt_l) {
    if (k < 20) return cnt_l * xT[l * CIN + k];
    if (k < 40) return accT[l * CIN + (k - 20)];
    return 0.f;
}

__global__ void k_hidden(const float* __restrict__ xT, const float* __restrict__ accT,
                         const float* __restrict__ cnt, const float* __restrict__ deg,
                         const _Float16* __restrict__ A1, _Float16* __restrict__ h) {
    int wave = threadIdx.x >> 5;
    int tile = blockIdx.x * 8 + wave;
    if (tile >= TILES_A) return;                 // wave-uniform: EXEC all-ones for WMMA
    int lane = threadIdx.x & 31;
    int half = lane >> 4;
    int n    = lane & 15;
    int l    = tile * 16 + n;

    float cnt_l = cnt[l];
    float deg_l = deg[l];
    float degc  = (deg_l == 0.f) ? 1.f : deg_l;
    float rdeg  = __builtin_amdgcn_rcpf(degc);   // one v_rcp_f32, degc >= 1

    // B fragments (K=0..31 and K=32..63), §7.12.2 16-bit layout
    v16h b0, b1;
#pragma unroll
    for (int i = 0; i < 8; ++i) {
        int k0 = half * 8 + i;        // half0: 0..7   half1: 8..15
        int k1 = 16 + half * 8 + i;   // half0: 16..23 half1: 24..31
        b0[i]     = (_Float16)bvalA(k0,      xT, accT, l, cnt_l);
        b0[i + 8] = (_Float16)bvalA(k1,      xT, accT, l, cnt_l);
        b1[i]     = (_Float16)bvalA(32 + k0, xT, accT, l, cnt_l);
        b1[i + 8] = (_Float16)bvalA(32 + k1, xT, accT, l, cnt_l);
    }

    // preload all A fragments (4 M-tiles x 2 K-steps) so loads overlap,
    // then issue the 8 WMMAs back-to-back
    v16h a0[4], a1[4];
#pragma unroll
    for (int mt = 0; mt < 4; ++mt) {
        const _Float16* Ar = A1 + (mt * 16 + n) * 64;
#pragma unroll
        for (int i = 0; i < 8; ++i) {
            a0[mt][i]     = Ar[half * 8 + i];
            a0[mt][i + 8] = Ar[16 + half * 8 + i];
            a1[mt][i]     = Ar[32 + half * 8 + i];
            a1[mt][i + 8] = Ar[48 + half * 8 + i];
        }
    }

    v8f acc[4];
#pragma unroll
    for (int mt = 0; mt < 4; ++mt) {
        v8f c = {};
        c = __builtin_amdgcn_wmma_f32_16x16x32_f16(false, a0[mt], false, b0, (short)0, c, false, false);
        c = __builtin_amdgcn_wmma_f32_16x16x32_f16(false, a1[mt], false, b1, (short)0, c, false, false);
        acc[mt] = c;
    }

#pragma unroll
    for (int mt = 0; mt < 4; ++mt) {
#pragma unroll
        for (int r = 0; r < 8; ++r) {
            int ch = mt * 16 + half * 8 + r;     // D layout: VGPR r, halves hold M=r / M=r+8
            float v;
            if (ch == 63) v = deg_l;             // counts channel (relu(deg)=deg)
            else          v = fmaxf(acc[mt][r] * rdeg, 0.f);
            h[ch * LP + 1 + l] = (_Float16)v;    // +1: left zero pad
        }
    }
}

// ---------------- stage B: strided conv as GEMM + per-tile max ----------------
__global__ void k_conv(const _Float16* __restrict__ h, const _Float16* __restrict__ A2,
                       float* __restrict__ partials) {
    int wave = threadIdx.x >> 5;
    int tile = blockIdx.x * 8 + wave;
    if (tile >= TILES_C) return;                 // wave-uniform
    int lane = threadIdx.x & 31;
    int half = lane >> 4;
    int n    = lane & 15;
    int t    = tile * 16 + n;                    // conv output position
    int tc   = t < LOUT ? t : (LOUT - 1);        // clamp loads; mask value later

    // 6 B fragments covering K=0..191; k = r*64+ic -> h[ic][2t + r] (padded index)
    v16h b[6];
#pragma unroll
    for (int ks = 0; ks < 6; ++ks) {
        int kb = ks * 32;
#pragma unroll
        for (int i = 0; i < 8; ++i) {
            int k0 = kb + half * 8 + i;
            int k1 = kb + 16 + half * 8 + i;
            b[ks][i]     = h[(k0 & 63) * LP + 2 * tc + (k0 >> 6)];
            b[ks][i + 8] = h[(k1 & 63) * LP + 2 * tc + (k1 >> 6)];
        }
    }

#pragma unroll
    for (int mt = 0; mt < 8; ++mt) {
        const _Float16* Ar = A2 + (mt * 16 + n) * 192;
        // load this M-tile's 6 A fragments first, then 6 back-to-back WMMAs
        v16h a[6];
#pragma unroll
        for (int ks = 0; ks < 6; ++ks) {
            int kb = ks * 32;
#pragma unroll
            for (int i = 0; i < 8; ++i) {
                a[ks][i]     = Ar[kb + half * 8 + i];
                a[ks][i + 8] = Ar[kb + 16 + half * 8 + i];
            }
        }
        v8f c = {};
#pragma unroll
        for (int ks = 0; ks < 6; ++ks)
            c = __builtin_amdgcn_wmma_f32_16x16x32_f16(false, a[ks], false, b[ks], (short)0, c, false, false);

        if (t >= LOUT) {
#pragma unroll
            for (int r = 0; r < 8; ++r) c[r] = -FLT_MAX;
        }
        // max over the 16 positions: xor-reduce within each 16-lane half
#pragma unroll
        for (int off = 8; off >= 1; off >>= 1)
#pragma unroll
            for (int r = 0; r < 8; ++r)
                c[r] = fmaxf(c[r], __shfl_xor(c[r], off, 32));
        if (n == 0) {
#pragma unroll
            for (int r = 0; r < 8; ++r)
                partials[tile * C2 + mt * 16 + half * 8 + r] = c[r];
        }
    }
}

// ---------------- final: max over tiles, +bias, dot w_lin, +b_lin ----------------
__global__ void k_final(const float* __restrict__ partials, const float* __restrict__ b_conv,
                        const float* __restrict__ w_lin, const float* __restrict__ b_lin,
                        float* __restrict__ out) {
    __shared__ float red[C2];
    int ch = threadIdx.x;
    float m = -FLT_MAX;
    for (int t = 0; t < TILES_C; ++t) m = fmaxf(m, partials[t * C2 + ch]);
    red[ch] = (m + b_conv[ch]) * w_lin[ch];
    __syncthreads();
    for (int s = 64; s > 0; s >>= 1) {
        if (ch < s) red[ch] += red[ch + s];
        __syncthreads();
    }
    if (ch == 0) out[0] = red[0] + b_lin[0];
}

extern "C" void kernel_launch(void* const* d_in, const int* in_sizes, int n_in,
                              void* d_out, int out_size, void* d_ws, size_t ws_size,
                              hipStream_t stream) {
    const float* seq    = (const float*)d_in[0];
    const int*   src    = (const int*)d_in[1];
    const int*   dst    = (const int*)d_in[2];
    const float* w_pair = (const float*)d_in[3];
    const float* w_conv = (const float*)d_in[4];
    const float* b_conv = (const float*)d_in[5];
    const float* w_lin  = (const float*)d_in[6];
    const float* b_lin  = (const float*)d_in[7];
    float* out = (float*)d_out;

    char* ws = (char*)d_ws;
    float*     accT = (float*)(ws + WS_ACC);
    float*     cnt  = (float*)(ws + WS_CNT);
    float*     deg  = (float*)(ws + WS_DEG);
    _Float16*  h    = (_Float16*)(ws + WS_H);
    float*     xT   = (float*)(ws + WS_XT);
    _Float16*  A1   = (_Float16*)(ws + WS_A1);
    _Float16*  A2   = (_Float16*)(ws + WS_A2);
    float*     part = (float*)(ws + WS_PART);

    k_zero<<<2048, 256, 0, stream>>>((unsigned*)ws, (int)WS_ZERO_DWORDS);
    k_transpose<<<(CIN * L_SEQ + 255) / 256, 256, 0, stream>>>(seq, xT);
    k_prep<<<(64 * 64 + 128 * 192 + 255) / 256, 256, 0, stream>>>(w_pair, w_conv, A1, A2);
    k_edge<<<(E_EDGES * CIN + 255) / 256, 256, 0, stream>>>(src, dst, xT, accT, cnt, deg);
    k_hidden<<<(TILES_A + 7) / 8, 256, 0, stream>>>(xT, accT, cnt, deg, A1, h);
    k_conv<<<(TILES_C + 7) / 8, 256, 0, stream>>>(h, A2, part);
    k_final<<<1, C2, 0, stream>>>(part, b_conv, w_lin, b_lin, out);
}
